// JTMPN_14946486190822
// MI455X (gfx1250) — matrix operand order repaired and checked by reading, not compiled, not run on hardware
//
#include <hip/hip_runtime.h>

typedef __attribute__((ext_vector_type(16))) _Float16 v16h;
typedef __attribute__((ext_vector_type(8)))  _Float16 v8h;
typedef __attribute__((ext_vector_type(8)))  float    v8f;

constexpr int ATOM_FDIM = 35;
constexpr int IN_DIM    = 40;                 // ATOM_FDIM + BOND_FDIM
constexpr int MAX_NB    = 15;
constexpr int HIDDEN    = 512;
constexpr int DEPTH     = 6;
constexpr int N_MOLS    = 1024;
constexpr int APM       = 32;
constexpr int N_ATOMS   = N_MOLS * APM;       // 32768
constexpr int N_BONDS   = 2 * N_ATOMS;        // 65536
constexpr int N_MESS    = 8192;
constexpr int TOTAL_MSG = N_MESS + N_BONDS;   // 73728

constexpr int KI = 64;    // IN_DIM padded to WMMA K granularity
constexpr int KO = 576;   // (ATOM_FDIM + HIDDEN)=547 padded

// ---------------- WMMA helpers ----------------

static __device__ __forceinline__ v8f wmma_f16(v16h a, v16h b, v8f c) {
  // D = A(16x32 f16) * B(32x16 f16) + C(16x16 f32)
  return __builtin_amdgcn_wmma_f32_16x16x32_f16(false, a, false, b, (short)0, c,
                                                false, false);
}

// A fragment (16x32 f16) from an LDS row-major tile. ISA 7.12.2:
// lane<16: row M=lane,   K = {k0..k0+7, k0+16..k0+23}
// lane>=16: row M=lane-16, K = {k0+8..k0+15, k0+24..k0+31}
static __device__ __forceinline__ v16h load_a_lds(const _Float16* lds, int pitch,
                                                  int mbase, int k0) {
  const int lane = threadIdx.x & 31;
  const int row  = mbase + (lane & 15);
  const int off  = (lane < 16) ? 0 : 8;
  const _Float16* p = lds + row * pitch + k0 + off;
  union { v16h v; v8h h[2]; } u;
  u.h[0] = *(const v8h*)(p);
  u.h[1] = *(const v8h*)(p + 16);
  return u.v;
}

// B fragment (32x16 f16) from pre-transposed weights WT[N][ldk] (row-major over N).
// lane<16: column N=n0+lane,    K = {k0..k0+7, k0+16..k0+23}
// lane>=16: column N=n0+lane-16, K = {k0+8..k0+15, k0+24..k0+31}
static __device__ __forceinline__ v16h load_b_glb(const _Float16* WT, int ldk,
                                                  int n0, int k0) {
  const int lane = threadIdx.x & 31;
  const int col  = n0 + (lane & 15);
  const int koff = (lane < 16) ? 0 : 8;
  const _Float16* p = WT + (size_t)col * ldk + k0 + koff;
  union { v16h v; v8h h[2]; } u;
  u.h[0] = *(const v8h*)(p);
  u.h[1] = *(const v8h*)(p + 16);
  return u.v;
}

// ---------------- prep kernels ----------------

// Transpose + f16-convert + zero-pad all three weight matrices.
__global__ __launch_bounds__(256) void prep_weights(
    const float* __restrict__ W_i, const float* __restrict__ W_h,
    const float* __restrict__ W_o, _Float16* __restrict__ WiT,
    _Float16* __restrict__ WhT, _Float16* __restrict__ WoT) {
  const int id = blockIdx.x * 256 + threadIdx.x;
  if (id >= HIDDEN * KO) return;
  const int n = id / KO, k = id % KO;
  WoT[(size_t)n * KO + k] =
      (_Float16)((k < ATOM_FDIM + HIDDEN) ? W_o[(size_t)k * HIDDEN + n] : 0.f);
  if (k < KI)
    WiT[(size_t)n * KI + k] =
        (_Float16)((k < IN_DIM) ? W_i[(size_t)k * HIDDEN + n] : 0.f);
  if (k < HIDDEN)
    WhT[(size_t)n * HIDDEN + k] = (_Float16)W_h[(size_t)k * HIDDEN + n];
}

// tree_message (f32) -> f16 head of both ping-pong message tables.
__global__ __launch_bounds__(256) void prep_tree(
    const float* __restrict__ tree, _Float16* __restrict__ msgA,
    _Float16* __restrict__ msgB) {
  const size_t id = (size_t)blockIdx.x * 256 + threadIdx.x;  // < N_MESS*HIDDEN
  const _Float16 v = (_Float16)tree[id];
  msgA[id] = v;
  msgB[id] = v;
}

// ---------------- binput = fbonds @ W_i ; graph_message = relu(binput) ----------------

__global__ __launch_bounds__(256) void binput_kernel(
    const float* __restrict__ fbonds, const _Float16* __restrict__ WiT,
    float* __restrict__ binput, _Float16* __restrict__ msg_out) {
  __shared__ _Float16 ldsA[32 * 72];  // 32 rows x KI, pitch 72 (16B aligned rows)
  const int tid = threadIdx.x, wave = tid >> 5, lane = tid & 31;
  const int b0 = blockIdx.x * 32;

  for (int idx = tid; idx < 32 * KI; idx += 256) {
    const int r = idx >> 6, k = idx & 63;
    const float v = (k < IN_DIM) ? fbonds[(size_t)(b0 + r) * IN_DIM + k] : 0.f;
    ldsA[r * 72 + k] = (_Float16)v;
  }
  __syncthreads();

  const int n0 = wave * 64;
  v8f c[2][4] = {};
  for (int k0 = 0; k0 < KI; k0 += 32) {
    const v16h a0 = load_a_lds(ldsA, 72, 0, k0);
    const v16h a1 = load_a_lds(ldsA, 72, 16, k0);
#pragma unroll
    for (int j = 0; j < 4; ++j) {
      const v16h b = load_b_glb(WiT, KI, n0 + j * 16, k0);
      c[0][j] = wmma_f16(a0, b, c[0][j]);
      c[1][j] = wmma_f16(a1, b, c[1][j]);
    }
  }

#pragma unroll
  for (int j = 0; j < 4; ++j) {
    const int col = n0 + j * 16 + (lane & 15);
#pragma unroll
    for (int i = 0; i < 2; ++i) {
#pragma unroll
      for (int g = 0; g < 8; ++g) {
        const int bond = b0 + i * 16 + ((lane < 16) ? g : g + 8);
        const float v = c[i][j][g];
        binput[(size_t)bond * HIDDEN + col] = v;
        msg_out[(size_t)(N_MESS + bond) * HIDDEN + col] =
            (_Float16)(v > 0.f ? v : 0.f);
      }
    }
  }
}

// ---------------- fused gather-sum + (nei @ W_h) + relu(binput + .) ----------------

__global__ __launch_bounds__(256) void mp_step_kernel(
    const _Float16* __restrict__ msg_in, _Float16* __restrict__ msg_out,
    const float* __restrict__ binput, const int* __restrict__ bgraph,
    const _Float16* __restrict__ WhT) {
  __shared__ _Float16 ldsA[32 * 520];  // nei tile, f16, padded pitch
  const int tid = threadIdx.x, wave = tid >> 5, lane = tid & 31;
  const int b0 = blockIdx.x * 32;

  // gather-sum over 15 neighbors (f16 table, f32 accumulation), L2-resident
  for (int rr = 0; rr < 4; ++rr) {
    const int r = wave * 4 + rr;
    const int bond = b0 + r;
    const int* gidx = bgraph + (size_t)bond * MAX_NB;
    float acc[16];
#pragma unroll
    for (int e = 0; e < 16; ++e) acc[e] = 0.f;
    for (int j = 0; j < MAX_NB; ++j) {
      const _Float16* src = msg_in + (size_t)gidx[j] * HIDDEN + lane * 16;
      const v8h x0 = *(const v8h*)(src);
      const v8h x1 = *(const v8h*)(src + 8);
#pragma unroll
      for (int e = 0; e < 8; ++e) {
        acc[e] += (float)x0[e];
        acc[8 + e] += (float)x1[e];
      }
    }
    v8h o0, o1;
#pragma unroll
    for (int e = 0; e < 8; ++e) {
      o0[e] = (_Float16)acc[e];
      o1[e] = (_Float16)acc[8 + e];
    }
    _Float16* dst = ldsA + r * 520 + lane * 16;
    *(v8h*)(dst) = o0;
    *(v8h*)(dst + 8) = o1;
  }
  __syncthreads();

  // [32 x 512] x [512 x 512] GEMM, wave owns 32x64 slab
  const int n0 = wave * 64;
  v8f c[2][4] = {};
  for (int k0 = 0; k0 < HIDDEN; k0 += 32) {
    const v16h a0 = load_a_lds(ldsA, 520, 0, k0);
    const v16h a1 = load_a_lds(ldsA, 520, 16, k0);
#pragma unroll
    for (int j = 0; j < 4; ++j) {
      const v16h b = load_b_glb(WhT, HIDDEN, n0 + j * 16, k0);
      c[0][j] = wmma_f16(a0, b, c[0][j]);
      c[1][j] = wmma_f16(a1, b, c[1][j]);
    }
  }

  // epilogue: relu(binput + acc) -> next message table (f16)
#pragma unroll
  for (int j = 0; j < 4; ++j) {
    const int col = n0 + j * 16 + (lane & 15);
#pragma unroll
    for (int i = 0; i < 2; ++i) {
#pragma unroll
      for (int g = 0; g < 8; ++g) {
        const int bond = b0 + i * 16 + ((lane < 16) ? g : g + 8);
        float v = c[i][j][g] + binput[(size_t)bond * HIDDEN + col];
        v = v > 0.f ? v : 0.f;
        msg_out[(size_t)(N_MESS + bond) * HIDDEN + col] = (_Float16)v;
      }
    }
  }
}

// ---------------- atom gather + W_o GEMM + bias/relu + per-mol mean ----------------

__global__ __launch_bounds__(256) void atom_kernel(
    const _Float16* __restrict__ msg, const int* __restrict__ agraph,
    const float* __restrict__ fatoms, const _Float16* __restrict__ WoT,
    const float* __restrict__ b_o, const int* __restrict__ scope,
    float* __restrict__ out) {
  __shared__ _Float16 ldsA[32 * 584];  // ainput tile: [fatoms | nei | pad], f16
  __shared__ float red[HIDDEN];
  const int tid = threadIdx.x, wave = tid >> 5, lane = tid & 31;
  const int mol = blockIdx.x;
  const int ab = mol * APM;  // one workgroup == one molecule (32 atoms)

  for (int idx = tid; idx < 32 * 584; idx += 256) ldsA[idx] = (_Float16)0.f;
  for (int cc = tid; cc < HIDDEN; cc += 256) red[cc] = 0.f;
  __syncthreads();

  // fatoms -> cols [0, 35)
  for (int idx = tid; idx < 32 * ATOM_FDIM; idx += 256) {
    const int r = idx / ATOM_FDIM, cc = idx % ATOM_FDIM;
    ldsA[r * 584 + cc] = (_Float16)fatoms[(size_t)(ab + r) * ATOM_FDIM + cc];
  }
  // nei gather-sum -> cols [35, 547)
  for (int rr = 0; rr < 4; ++rr) {
    const int r = wave * 4 + rr;
    const int atom = ab + r;
    const int* gidx = agraph + (size_t)atom * MAX_NB;
    float acc[16];
#pragma unroll
    for (int e = 0; e < 16; ++e) acc[e] = 0.f;
    for (int j = 0; j < MAX_NB; ++j) {
      const _Float16* src = msg + (size_t)gidx[j] * HIDDEN + lane * 16;
      const v8h x0 = *(const v8h*)(src);
      const v8h x1 = *(const v8h*)(src + 8);
#pragma unroll
      for (int e = 0; e < 8; ++e) {
        acc[e] += (float)x0[e];
        acc[8 + e] += (float)x1[e];
      }
    }
#pragma unroll
    for (int e = 0; e < 16; ++e)
      ldsA[r * 584 + ATOM_FDIM + lane * 16 + e] = (_Float16)acc[e];
  }
  __syncthreads();

  // [32 x 576] x [576 x 512] GEMM
  const int n0 = wave * 64;
  v8f c[2][4] = {};
  for (int k0 = 0; k0 < KO; k0 += 32) {
    const v16h a0 = load_a_lds(ldsA, 584, 0, k0);
    const v16h a1 = load_a_lds(ldsA, 584, 16, k0);
#pragma unroll
    for (int j = 0; j < 4; ++j) {
      const v16h b = load_b_glb(WoT, KO, n0 + j * 16, k0);
      c[0][j] = wmma_f16(a0, b, c[0][j]);
      c[1][j] = wmma_f16(a1, b, c[1][j]);
    }
  }

  // bias + relu, reduce the 32 atom rows into per-column sums
#pragma unroll
  for (int j = 0; j < 4; ++j) {
    const int nc = n0 + j * 16 + (lane & 15);
    const float bias = b_o[nc];
    float part = 0.f;
#pragma unroll
    for (int i = 0; i < 2; ++i) {
#pragma unroll
      for (int g = 0; g < 8; ++g) {
        const float v = c[i][j][g] + bias;
        part += (v > 0.f) ? v : 0.f;
      }
    }
    atomicAdd(&red[nc], part);
  }
  __syncthreads();

  const float inv = 1.0f / (float)scope[mol * 2 + 1];
  for (int cc = tid; cc < HIDDEN; cc += 256)
    out[(size_t)mol * HIDDEN + cc] = red[cc] * inv;
}

// ---------------- launcher ----------------

extern "C" void kernel_launch(void* const* d_in, const int* in_sizes, int n_in,
                              void* d_out, int out_size, void* d_ws, size_t ws_size,
                              hipStream_t stream) {
  const float* fatoms = (const float*)d_in[0];
  const float* fbonds = (const float*)d_in[1];
  const int*   agraph = (const int*)d_in[2];
  const int*   bgraph = (const int*)d_in[3];
  const int*   scope  = (const int*)d_in[4];
  const float* tree   = (const float*)d_in[5];
  const float* W_i    = (const float*)d_in[6];
  const float* W_h    = (const float*)d_in[7];
  const float* W_o    = (const float*)d_in[8];
  const float* b_o    = (const float*)d_in[9];
  float* out = (float*)d_out;
  (void)in_sizes; (void)n_in; (void)out_size; (void)ws_size;

  char* ws = (char*)d_ws;
  size_t off = 0;
  auto carve = [&](size_t bytes) -> void* {
    void* p = ws + off;
    off += (bytes + 255) & ~(size_t)255;
    return p;
  };
  _Float16* msgA   = (_Float16*)carve((size_t)TOTAL_MSG * HIDDEN * 2);  // 75.5 MB
  _Float16* msgB   = (_Float16*)carve((size_t)TOTAL_MSG * HIDDEN * 2);  // 75.5 MB
  float*    binput = (float*)   carve((size_t)N_BONDS * HIDDEN * 4);    // 128 MB
  _Float16* WiT    = (_Float16*)carve((size_t)HIDDEN * KI * 2);
  _Float16* WhT    = (_Float16*)carve((size_t)HIDDEN * HIDDEN * 2);
  _Float16* WoT    = (_Float16*)carve((size_t)HIDDEN * KO * 2);

  prep_weights<<<(HIDDEN * KO + 255) / 256, 256, 0, stream>>>(W_i, W_h, W_o,
                                                              WiT, WhT, WoT);
  prep_tree<<<(N_MESS * HIDDEN) / 256, 256, 0, stream>>>(tree, msgA, msgB);
  binput_kernel<<<N_BONDS / 32, 256, 0, stream>>>(fbonds, WiT, binput, msgA);

  _Float16* cur = msgA;
  _Float16* nxt = msgB;
  for (int d = 0; d < DEPTH - 1; ++d) {
    mp_step_kernel<<<N_BONDS / 32, 256, 0, stream>>>(cur, nxt, binput, bgraph, WhT);
    _Float16* t = cur; cur = nxt; nxt = t;
  }

  atom_kernel<<<N_MOLS, 256, 0, stream>>>(cur, agraph, fatoms, WoT, b_o, scope, out);
}
